// CrossAttentionModule_15109694947938
// MI455X (gfx1250) — matrix operand order repaired and checked by reading, not compiled, not run on hardware
//
#include <hip/hip_runtime.h>
#include <hip/hip_bf16.h>
#include <cstddef>
#include <cstdint>

// ---------------------------------------------------------------------------
// MI455X (gfx1250) fused cross-attention:
//   3x GEMM (32768x512x512) -> per-token 16x16 head-mix attention -> 1x GEMM
// GEMMs: v_wmma_f32_16x16x32_f16, double-buffered LDS tiles (1 barrier/K-step),
//        register-staged global fetch overlapped with WMMA.
// Attention: q/k/v rows staged via global_load_async_to_lds_b128 (ASYNCcnt).
// Workspace layout (f16): q | k | v | attn_out, 32 MB each.
// ---------------------------------------------------------------------------

typedef __attribute__((ext_vector_type(16))) _Float16 v16h;
typedef __attribute__((ext_vector_type(8)))  _Float16 v8h;
typedef __attribute__((ext_vector_type(4)))  _Float16 v4h;
typedef __attribute__((ext_vector_type(8)))  float    v8f;

template <typename T> struct Vec4;
template <> struct Vec4<float>    { using type = float4; };
template <> struct Vec4<_Float16> { using type = v4h;    };

__device__ __forceinline__ void storeC(float* p, float v)    { *p = v; }
__device__ __forceinline__ void storeC(_Float16* p, float v) { *p = (_Float16)v; }

// LDS byte offset of a generic pointer into shared memory (for async-to-LDS VDST)
__device__ __forceinline__ unsigned lds_addr(const void* p) {
  return (unsigned)(unsigned long long)(__attribute__((address_space(3))) const char*)p;
}

// ---------------------------------------------------------------------------
// C[M,512] = A[M,512] @ W[512,512] + bias   (row-major, W stored [in,out])
// Block: 256 threads (8 waves). Block tile 128x128, K-step 32.
// Wave grid 4(M) x 2(N): each wave owns 2x4 16x16 f32 accumulators.
// ---------------------------------------------------------------------------
template <typename TIn, typename TOut>
__global__ __launch_bounds__(256) void gemm512_wmma(
    const TIn* __restrict__ A, const float* __restrict__ W,
    const float* __restrict__ bias, TOut* __restrict__ C, int M) {
  constexpr int K = 512, NOUT = 512;
  constexpr int BM = 128, BN = 128, BK = 32;
  constexpr int AP = BK + 8;  // LDS row pitch (halves): 80B, 16B aligned
  constexpr int BP = BK + 8;
  constexpr int KSTEPS = K / BK;

  __shared__ __align__(16) _Float16 As[2][BM * AP];  // A tiles, row-major [m][k]
  __shared__ __align__(16) _Float16 Bt[2][BN * BP];  // W tiles, transposed [n][k]

  const int tid   = threadIdx.x;
  const int lane  = tid & 31;
  const int wave  = tid >> 5;
  const int waveM = wave & 3;   // 4 waves along M (32 rows each)
  const int waveN = wave >> 2;  // 2 waves along N (64 cols each)
  const int m0    = blockIdx.y * BM;
  const int n0    = blockIdx.x * BN;

  const int h  = lane >> 4;     // lane half (ISA A/B/C layouts)
  const int l4 = lane & 15;

  using AV4 = typename Vec4<TIn>::type;
  AV4    aReg[4];
  float4 wReg[4];

  auto fetch = [&](int k0) {
#pragma unroll
    for (int i = 0; i < 4; ++i) {
      const int idx = tid + i * 256;         // 0..1023
      const int row = idx >> 3;              // 0..127
      const int c4  = (idx & 7) * 4;         // 0..28
      aReg[i] = *reinterpret_cast<const AV4*>(A + (size_t)(m0 + row) * K + k0 + c4);
    }
#pragma unroll
    for (int i = 0; i < 4; ++i) {
      const int idx = tid + i * 256;
      const int kk  = idx >> 5;              // 0..31
      const int n4  = (idx & 31) * 4;        // 0..124
      wReg[i] = *reinterpret_cast<const float4*>(W + (size_t)(k0 + kk) * NOUT + n0 + n4);
    }
  };
  auto stage = [&](int buf) {
#pragma unroll
    for (int i = 0; i < 4; ++i) {
      const int idx = tid + i * 256;
      const int row = idx >> 3;
      const int c4  = (idx & 7) * 4;
      _Float16* d = &As[buf][row * AP + c4];
      d[0] = (_Float16)aReg[i].x; d[1] = (_Float16)aReg[i].y;
      d[2] = (_Float16)aReg[i].z; d[3] = (_Float16)aReg[i].w;
    }
#pragma unroll
    for (int i = 0; i < 4; ++i) {
      const int idx = tid + i * 256;
      const int kk  = idx >> 5;
      const int n4  = (idx & 31) * 4;
      Bt[buf][(n4 + 0) * BP + kk] = (_Float16)wReg[i].x;
      Bt[buf][(n4 + 1) * BP + kk] = (_Float16)wReg[i].y;
      Bt[buf][(n4 + 2) * BP + kk] = (_Float16)wReg[i].z;
      Bt[buf][(n4 + 3) * BP + kk] = (_Float16)wReg[i].w;
    }
  };

  v8f acc[2][4] = {};

  fetch(0);
  stage(0);
  __syncthreads();

  int buf = 0;
  for (int kt = 0; kt < KSTEPS; ++kt) {
    // overlap: pull tile kt+1 into VGPRs while WMMAs chew on tile kt
    if (kt + 1 < KSTEPS) {
      fetch((kt + 1) * BK);
      if (kt + 2 < KSTEPS)  // warm L2 two tiles ahead (global_prefetch_b8)
        __builtin_prefetch(W + (size_t)((kt + 2) * BK + (tid >> 3)) * NOUT + n0, 0, 1);
    }

    // -- fragments per ISA 16-bit layouts ------------------------------------
    // A 16x32: lane half h holds K in {8h..8h+7} U {16+8h..16+8h+7}
    v16h afrag[2];
#pragma unroll
    for (int mi = 0; mi < 2; ++mi) {
      const _Float16* p = &As[buf][(waveM * 32 + mi * 16 + l4) * AP + h * 8];
      const v8h lo = *reinterpret_cast<const v8h*>(p);
      const v8h hi = *reinterpret_cast<const v8h*>(p + 16);
      v16h a;
#pragma unroll
      for (int e = 0; e < 8; ++e) { a[e] = lo[e]; a[e + 8] = hi[e]; }
      afrag[mi] = a;
    }
    // B 32x16: lane = column (lane&15), K in {8h..8h+7} U {16+8h..16+8h+7}
    v16h bfrag[4];
#pragma unroll
    for (int ni = 0; ni < 4; ++ni) {
      const _Float16* p = &Bt[buf][(waveN * 64 + ni * 16 + l4) * BP + h * 8];
      const v8h lo = *reinterpret_cast<const v8h*>(p);
      const v8h hi = *reinterpret_cast<const v8h*>(p + 16);
      v16h b;
#pragma unroll
      for (int e = 0; e < 8; ++e) { b[e] = lo[e]; b[e + 8] = hi[e]; }
      bfrag[ni] = b;
    }

    // -- 8 WMMAs per wave per K-step -----------------------------------------
#pragma unroll
    for (int mi = 0; mi < 2; ++mi)
#pragma unroll
      for (int ni = 0; ni < 4; ++ni)
        acc[mi][ni] = __builtin_amdgcn_wmma_f32_16x16x32_f16(
            false, afrag[mi], false, bfrag[ni], (short)0, acc[mi][ni],
            false, false);

    // -- stage tile kt+1 into the alternate buffer; one barrier per K-step ---
    if (kt + 1 < KSTEPS) {
      stage(buf ^ 1);
      __syncthreads();   // waves pass only after their DS reads+writes drain
      buf ^= 1;
    }
  }

  // -- epilogue: C/D layout -> row = r + 8h, col = lane&15 --------------------
#pragma unroll
  for (int mi = 0; mi < 2; ++mi) {
#pragma unroll
    for (int ni = 0; ni < 4; ++ni) {
      const int col = n0 + waveN * 64 + ni * 16 + l4;
      const float b = bias[col];
#pragma unroll
      for (int r = 0; r < 8; ++r) {
        const int row = m0 + waveM * 32 + mi * 16 + r + 8 * h;
        storeC(&C[(size_t)row * NOUT + col], acc[mi][ni][r] + b);
      }
    }
  }
}

// ---------------------------------------------------------------------------
// Per-token head-mixing attention: q,k,v rows [16 heads x 32 dh] (f16).
// One wave per token (8 tokens / 256-thread block). q/k/v rows are staged to
// LDS with gfx1250 async-to-LDS copies (ASYNCcnt), then f32 VALU math with
// __shfl_xor(.,16) pair reductions for the 16-wide softmax rows.
// ---------------------------------------------------------------------------
__global__ __launch_bounds__(256) void attn_headmix(
    const _Float16* __restrict__ q, const _Float16* __restrict__ k,
    const _Float16* __restrict__ v, _Float16* __restrict__ out, int T) {
  constexpr int H = 16, DH = 32, CO = 512;
  const float scale = 0.17677669529663687f;  // 1/sqrt(32)

  __shared__ __align__(16) _Float16 sm[8 * 3 * CO];  // 24 KB
  const int wave = threadIdx.x >> 5;
  const int lane = threadIdx.x & 31;
  const int token = blockIdx.x * 8 + wave;
  if (token >= T) return;

  _Float16* sq = &sm[wave * 3 * CO];
  _Float16* sk = sq + CO;
  _Float16* sv = sk + CO;

  // -- async stage: each lane moves 32B of each of q,k,v straight to LDS -----
  {
    const unsigned l0 = lds_addr(sq) + lane * 32;
    const unsigned l1 = lds_addr(sk) + lane * 32;
    const unsigned l2 = lds_addr(sv) + lane * 32;
    const unsigned long long g0 = (unsigned long long)(q + (size_t)token * CO + lane * 16);
    const unsigned long long g1 = (unsigned long long)(k + (size_t)token * CO + lane * 16);
    const unsigned long long g2 = (unsigned long long)(v + (size_t)token * CO + lane * 16);
    // instruction offset applies to both LDS dest and global source addresses
    asm volatile("global_load_async_to_lds_b128 %0, %1, off"           :: "v"(l0), "v"(g0) : "memory");
    asm volatile("global_load_async_to_lds_b128 %0, %1, off offset:16" :: "v"(l0), "v"(g0) : "memory");
    asm volatile("global_load_async_to_lds_b128 %0, %1, off"           :: "v"(l1), "v"(g1) : "memory");
    asm volatile("global_load_async_to_lds_b128 %0, %1, off offset:16" :: "v"(l1), "v"(g1) : "memory");
    asm volatile("global_load_async_to_lds_b128 %0, %1, off"           :: "v"(l2), "v"(g2) : "memory");
    asm volatile("global_load_async_to_lds_b128 %0, %1, off offset:16" :: "v"(l2), "v"(g2) : "memory");
    asm volatile("s_wait_asynccnt 0x0" ::: "memory");
  }
  // wave-private LDS region: no cross-wave barrier needed

  const int h = lane & 15;
  const int half = lane >> 4;

  float qr[DH];
#pragma unroll
  for (int d = 0; d < DH; ++d) qr[d] = (float)sq[h * DH + d];

  // scores for 8 key-heads
  float s[8];
#pragma unroll
  for (int j = 0; j < 8; ++j) {
    const int g = half * 8 + j;
    float a = 0.f;
#pragma unroll
    for (int d = 0; d < DH; ++d) a += qr[d] * (float)sk[g * DH + d];
    s[j] = a * scale;
  }
  // row softmax across lane pair
  float mx = s[0];
#pragma unroll
  for (int j = 1; j < 8; ++j) mx = fmaxf(mx, s[j]);
  mx = fmaxf(mx, __shfl_xor(mx, 16, 32));
  float sum = 0.f;
#pragma unroll
  for (int j = 0; j < 8; ++j) { s[j] = __expf(s[j] - mx); sum += s[j]; }
  sum += __shfl_xor(sum, 16, 32);
  const float inv = 1.f / sum;

  // gather full attn row (16 weights) on each lane of the pair
  float attn[16];
#pragma unroll
  for (int j = 0; j < 8; ++j) {
    attn[half * 8 + j]       = s[j] * inv;
    attn[(half ^ 1) * 8 + j] = __shfl_xor(s[j], 16, 32) * inv;
  }

  // out[h][d], d in [16*half, 16*half+16)
  float o[16];
#pragma unroll
  for (int d = 0; d < 16; ++d) o[d] = 0.f;
#pragma unroll
  for (int g = 0; g < H; ++g) {
    const float ag = attn[g];
    const _Float16* vp = &sv[g * DH + half * 16];
#pragma unroll
    for (int d = 0; d < 16; ++d) o[d] += ag * (float)vp[d];
  }

  v8h o0, o1;
#pragma unroll
  for (int d = 0; d < 8; ++d) { o0[d] = (_Float16)o[d]; o1[d] = (_Float16)o[d + 8]; }
  _Float16* gout = out + (size_t)token * CO + h * DH + half * 16;
  *reinterpret_cast<v8h*>(gout)     = o0;
  *reinterpret_cast<v8h*>(gout + 8) = o1;
}

// ---------------------------------------------------------------------------
extern "C" void kernel_launch(void* const* d_in, const int* in_sizes, int n_in,
                              void* d_out, int out_size, void* d_ws, size_t ws_size,
                              hipStream_t stream) {
  const float* query = (const float*)d_in[0];
  const float* key   = (const float*)d_in[1];
  const float* value = (const float*)d_in[2];
  const float* Wq = (const float*)d_in[3];
  const float* bq = (const float*)d_in[4];
  const float* Wk = (const float*)d_in[5];
  const float* bk = (const float*)d_in[6];
  const float* Wv = (const float*)d_in[7];
  const float* bv = (const float*)d_in[8];
  const float* Wo = (const float*)d_in[9];
  const float* bo = (const float*)d_in[10];

  const int C = 512;
  const int M = in_sizes[0] / C;  // B*N tokens (32768)

  const size_t planeBytes = (size_t)M * C * sizeof(_Float16);
  _Float16* qh = (_Float16*)d_ws;
  _Float16* kh = (_Float16*)((char*)d_ws + 1 * planeBytes);
  _Float16* vh = (_Float16*)((char*)d_ws + 2 * planeBytes);
  _Float16* ah = (_Float16*)((char*)d_ws + 3 * planeBytes);

  dim3 gGrid(C / 128, M / 128);  // (4, 256)
  dim3 blk(256);

  gemm512_wmma<float, _Float16><<<gGrid, blk, 0, stream>>>(query, Wq, bq, qh, M);
  gemm512_wmma<float, _Float16><<<gGrid, blk, 0, stream>>>(key,   Wk, bk, kh, M);
  gemm512_wmma<float, _Float16><<<gGrid, blk, 0, stream>>>(value, Wv, bv, vh, M);

  attn_headmix<<<dim3(M / 8), blk, 0, stream>>>(qh, kh, vh, ah, M);

  gemm512_wmma<_Float16, float><<<gGrid, blk, 0, stream>>>(ah, Wo, bo, (float*)d_out, M);
}
